// GenerativeModelPriorDistribution_85057532330137
// MI455X (gfx1250) — matrix operand order repaired and checked by reading, not compiled, not run on hardware
//
#include <hip/hip_runtime.h>

// ---------------------------------------------------------------------------
// 2-layer LSTM LM NLL on MI455X (gfx1250).
// bf16 WMMA + LDS-resident layer-1 weights (loaded once via TDM) +
// persistent cooperative kernel with 3 grid barriers per timestep.
// ---------------------------------------------------------------------------

#define B_   256
#define T_   800
#define DIN  128
#define H_   512
#define V_   128
#define G4H  2048          // 4*H
#define NBLK 64
#define NTHR 256

typedef __bf16 bf16;
typedef __attribute__((ext_vector_type(16))) __bf16 bf16x16;
typedef __attribute__((ext_vector_type(8)))  __bf16 bf16x8;
typedef __attribute__((ext_vector_type(8)))  float  f32x8;
typedef __attribute__((ext_vector_type(4)))  unsigned int u32x4;
typedef __attribute__((ext_vector_type(8)))  int i32x8;
typedef __attribute__((ext_vector_type(4)))  int i32x4;

// ---- WMMA fragment loaders (CDNA5 16x16x32 bf16 layouts, wave32) ----------
// A (16x32, MxK): lane<16 -> row=lane, K {0..7}+{16..23}; lane>=16 -> K {8..15}+{24..31}
__device__ __forceinline__ bf16x16 frag_a(const bf16* __restrict__ tile, int ld, int lane) {
  const int r  = lane & 15;
  const int kh = (lane >> 4) << 3;           // 0 or 8
  const bf16* p = tile + r * ld + kh;
  bf16x8 lo = *(const bf16x8*)(p);
  bf16x8 hi = *(const bf16x8*)(p + 16);
  return __builtin_shufflevector(lo, hi, 0,1,2,3,4,5,6,7,8,9,10,11,12,13,14,15);
}
// B (32x16, KxN): column n of B == row n of W (row-major, stride ld);
// lane<16 -> col=lane, K {0..7}+{16..23} ; lane>=16 -> col=lane-16, K {8..15}+{24..31}
__device__ __forceinline__ bf16x16 frag_b(const bf16* __restrict__ wrows, int ld, int lane) {
  const bf16* p = wrows + (size_t)(lane & 15) * ld + ((lane >> 4) << 3);
  bf16x8 lo = *(const bf16x8*)(p);
  bf16x8 hi = *(const bf16x8*)(p + 16);
  return __builtin_shufflevector(lo, hi, 0,1,2,3,4,5,6,7,8,9,10,11,12,13,14,15);
}

__device__ __forceinline__ f32x8 wmma_bf16(bf16x16 a, bf16x16 b, f32x8 c) {
  return __builtin_amdgcn_wmma_f32_16x16x32_bf16(false, a, false, b, (short)0, c,
                                                 false, false);
}

__device__ __forceinline__ float sigm(float x) { return 1.0f / (1.0f + __expf(-x)); }

// ---- device-wide sense barrier (persistent cooperative kernel) ------------
__device__ __forceinline__ void gridsync(int* cnt, int* gen, int& lgen) {
  __syncthreads();
  if (threadIdx.x == 0) {
    __threadfence();
    lgen += 1;
    if (atomicAdd(cnt, 1) == (int)gridDim.x - 1) {
      atomicExch(cnt, 0);
      __threadfence();
      atomicExch(gen, lgen);
    } else {
      while (atomicAdd(gen, 0) < lgen) __builtin_amdgcn_s_sleep(2);
    }
    __threadfence();
  }
  __syncthreads();
}

// ---------------------------------------------------------------------------
// Prepass kernels: weight reorder (gate-interleave) + bf16 convert + zeroing.
// Interleaved row index np = 4*j + gate  <->  original n = gate*H + j.
// ---------------------------------------------------------------------------
__global__ void k_zero32(int* __restrict__ p, int n) {
  int i = blockIdx.x * 256 + threadIdx.x;
  if (i < n) p[i] = 0;
}
__global__ void k_prep_wr0(const float* __restrict__ Whh0, bf16* __restrict__ Wr0) {
  int idx = blockIdx.x * 256 + threadIdx.x;            // 2048*512
  int np = idx >> 9, k = idx & 511;
  int n = (np & 3) * H_ + (np >> 2);
  Wr0[idx] = (bf16)Whh0[n * H_ + k];
}
__global__ void k_prep_wr1(const float* __restrict__ Wih1, const float* __restrict__ Whh1,
                           bf16* __restrict__ Wr1) {
  int idx = blockIdx.x * 256 + threadIdx.x;            // 2048*1024
  int np = idx >> 10, k = idx & 1023;
  int n = (np & 3) * H_ + (np >> 2);
  float v = (k < H_) ? Wih1[n * H_ + k] : Whh1[n * H_ + (k - H_)];
  Wr1[idx] = (bf16)v;
}
__global__ void k_prep_emb(const float* __restrict__ Wih0, const float* __restrict__ bih0,
                           const float* __restrict__ bhh0, float* __restrict__ emb0) {
  int idx = blockIdx.x * 256 + threadIdx.x;            // 129*2048
  if (idx >= (V_ + 1) * G4H) return;
  int v = idx >> 11, np = idx & 2047;
  int n = (np & 3) * H_ + (np >> 2);
  float val = bih0[n] + bhh0[n];
  if (v < V_) val += Wih0[n * DIN + v];
  emb0[idx] = val;
}
__global__ void k_prep_b1(const float* __restrict__ bih1, const float* __restrict__ bhh1,
                          float* __restrict__ bias1) {
  int np = blockIdx.x * 256 + threadIdx.x;             // 2048
  int n = (np & 3) * H_ + (np >> 2);
  bias1[np] = bih1[n] + bhh1[n];
}
__global__ void k_prep_wlin(const float* __restrict__ Wl, bf16* __restrict__ Wlin) {
  int idx = blockIdx.x * 256 + threadIdx.x;            // 128*512
  Wlin[idx] = (bf16)Wl[idx];
}

// ---------------------------------------------------------------------------
// Persistent kernel: 64 blocks x 256 threads = 512 waves.
// Block bx: jt = bx>>1 (hidden tile, shared by all 8 waves -> shared B panel),
//           mt = (bx&1)*8 + wave (batch tile).
// Layer-1 weight panel (64 rows x 1024 K, bf16, 128KB) resident in LDS.
// ---------------------------------------------------------------------------
__global__ __launch_bounds__(NTHR) void lstm_persistent(
    const bf16* __restrict__ Wr0, const bf16* __restrict__ Wr1,
    const bf16* __restrict__ Wlin, const float* __restrict__ emb0,
    const float* __restrict__ bias1, const float* __restrict__ blin,
    const int* __restrict__ z, const int* __restrict__ numf,
    bf16* __restrict__ h0, bf16* __restrict__ h1,
    float* __restrict__ c0, float* __restrict__ c1,
    float* __restrict__ nllacc, int* bar_cnt, int* bar_gen,
    float* __restrict__ out)
{
  __shared__ bf16  wr1s[64 * 1024];       // resident layer-1 weight panel (128KB)
  __shared__ float stage[8][16][17];      // per-wave 16x16 tile stage (padded)

  const int tid  = threadIdx.x;
  const int lane = tid & 31;
  const int w    = tid >> 5;
  const int wid  = blockIdx.x * (NTHR / 32) + w;     // 0..511
  const int hi   = lane >> 4;                        // half-wave
  const int cn   = lane & 15;
  const int jt   = blockIdx.x >> 1;                  // hidden tile 0..31 (per block)
  const int mt   = (blockIdx.x & 1) * 8 + w;         // batch tile 0..15 (per wave)
  const int BH   = B_ * H_;
  const f32x8 fzero = {0.f,0.f,0.f,0.f,0.f,0.f,0.f,0.f};
  int lgen = 0;

  // ---- one-time: stage this block's Wr1 panel (64 rows x 1024 K) into LDS --
  {
    const bf16* src = Wr1 + (size_t)jt * 64 * 1024;
#if __has_builtin(__builtin_amdgcn_tensor_load_to_lds)
    if (tid < 32) {
      unsigned lds_off = (unsigned)(size_t)(&wr1s[0]);
      unsigned long long ga = (unsigned long long)(size_t)src;
      // D# group0: count=1 | lds_addr | global_addr | type=2
      u32x4 g0 = { 1u, lds_off, (unsigned)ga,
                   ((unsigned)((ga >> 32) & 0x01FFFFFFull)) | 0x80000000u };
      // D# group1: data_size=2B; tensor_dim0=1024, tensor_dim1=64;
      //            tile_dim0=1024, tile_dim1=64; tensor_dim0_stride=1024
      i32x8 g1 = { (int)0x00010000u, (int)(1024u << 16), (int)(64u << 16),
                   (int)(1024u << 16), 64, 1024, 0, 0 };
      i32x4 gz  = { 0, 0, 0, 0 };
      i32x8 gz8 = { 0, 0, 0, 0, 0, 0, 0, 0 };
      __builtin_amdgcn_tensor_load_to_lds(g0, g1, gz, gz, gz8, 0);
      __builtin_amdgcn_s_wait_tensorcnt(0);
    }
#else
    for (int i = tid; i < (64 * 1024) / 8; i += NTHR)
      ((bf16x8*)wr1s)[i] = ((const bf16x8*)src)[i];
#endif
    __syncthreads();
  }

  for (int t = 0; t < T_; ++t) {
    const int par = t & 1;

    // ---------------- layer 0: gates = h0_old @ Wr0^T + emb[z] -------------
    {
      const bf16* A  = h0 + (size_t)par * BH + (size_t)mt * 16 * H_;
      const bf16* Bw = Wr0 + (size_t)(jt * 64) * H_;
      f32x8 acc[4];
#pragma unroll
      for (int nt = 0; nt < 4; ++nt) acc[nt] = fzero;
      for (int k = 0; k < H_; k += 32) {
        bf16x16 af = frag_a(A + k, H_, lane);
        acc[0] = wmma_bf16(af, frag_b(Bw +          k, H_, lane), acc[0]);
        acc[1] = wmma_bf16(af, frag_b(Bw + 16*H_ +  k, H_, lane), acc[1]);
        acc[2] = wmma_bf16(af, frag_b(Bw + 32*H_ +  k, H_, lane), acc[2]);
        acc[3] = wmma_bf16(af, frag_b(Bw + 48*H_ +  k, H_, lane), acc[3]);
      }
      // per-tile stage + cell update (wave-private LDS slot, no block sync)
      const int ul = lane & 3;                 // unit within tile
      const int rb = lane >> 2;                // 0..7
#pragma unroll
      for (int nt = 0; nt < 4; ++nt) {
#pragma unroll
        for (int r = 0; r < 8; ++r) stage[w][r + 8*hi][cn] = acc[nt][r];
        __builtin_amdgcn_wave_barrier();
        const int j  = jt * 16 + nt * 4 + ul;
        const int n0 = jt * 64 + (nt * 4 + ul) * 4;
#pragma unroll
        for (int rr = 0; rr < 2; ++rr) {
          const int row = rb + rr * 8;
          const int b   = mt * 16 + row;
          const int ev  = (t == 0) ? V_ : z[b * T_ + (t - 1)];
          const float4 e = *(const float4*)(emb0 + (size_t)ev * G4H + n0);
          float gi = stage[w][row][4*ul + 0] + e.x;
          float gf = stage[w][row][4*ul + 1] + e.y;
          float gg = stage[w][row][4*ul + 2] + e.z;
          float go = stage[w][row][4*ul + 3] + e.w;
          const int ci = b * H_ + j;
          float cv   = c0[ci];
          float cnew = sigm(gf) * cv + sigm(gi) * tanhf(gg);
          float hnew = sigm(go) * tanhf(cnew);
          c0[ci] = cnew;
          h0[(size_t)(par ^ 1) * BH + ci] = (bf16)hnew;
        }
        __builtin_amdgcn_wave_barrier();
      }
    }
    gridsync(bar_cnt, bar_gen, lgen);

    // ------- layer 1: gates = [h0_new, h1_old] @ Wr1^T + b1 (B from LDS) ---
    {
      const bf16* A0 = h0 + (size_t)(par ^ 1) * BH + (size_t)mt * 16 * H_;
      const bf16* A1 = h1 + (size_t)par * BH + (size_t)mt * 16 * H_;
      const bf16* Bl = wr1s;                         // LDS-resident panel
      f32x8 acc[4];
#pragma unroll
      for (int nt = 0; nt < 4; ++nt) acc[nt] = fzero;
      for (int k = 0; k < H_; k += 32) {             // K 0..511 : h0_new
        bf16x16 af = frag_a(A0 + k, H_, lane);
        acc[0] = wmma_bf16(af, frag_b(Bl +            k, 1024, lane), acc[0]);
        acc[1] = wmma_bf16(af, frag_b(Bl + 16*1024 +  k, 1024, lane), acc[1]);
        acc[2] = wmma_bf16(af, frag_b(Bl + 32*1024 +  k, 1024, lane), acc[2]);
        acc[3] = wmma_bf16(af, frag_b(Bl + 48*1024 +  k, 1024, lane), acc[3]);
      }
      for (int k = 0; k < H_; k += 32) {             // K 512..1023 : h1_old
        bf16x16 af = frag_a(A1 + k, H_, lane);
        acc[0] = wmma_bf16(af, frag_b(Bl + 512 +           k, 1024, lane), acc[0]);
        acc[1] = wmma_bf16(af, frag_b(Bl + 16*1024 + 512 + k, 1024, lane), acc[1]);
        acc[2] = wmma_bf16(af, frag_b(Bl + 32*1024 + 512 + k, 1024, lane), acc[2]);
        acc[3] = wmma_bf16(af, frag_b(Bl + 48*1024 + 512 + k, 1024, lane), acc[3]);
      }
      const int ul = lane & 3;
      const int rb = lane >> 2;
#pragma unroll
      for (int nt = 0; nt < 4; ++nt) {
#pragma unroll
        for (int r = 0; r < 8; ++r) stage[w][r + 8*hi][cn] = acc[nt][r];
        __builtin_amdgcn_wave_barrier();
        const int j  = jt * 16 + nt * 4 + ul;
        const int n0 = jt * 64 + (nt * 4 + ul) * 4;
        const float4 bb = *(const float4*)(bias1 + n0);
#pragma unroll
        for (int rr = 0; rr < 2; ++rr) {
          const int row = rb + rr * 8;
          const int b   = mt * 16 + row;
          float gi = stage[w][row][4*ul + 0] + bb.x;
          float gf = stage[w][row][4*ul + 1] + bb.y;
          float gg = stage[w][row][4*ul + 2] + bb.z;
          float go = stage[w][row][4*ul + 3] + bb.w;
          const int ci = b * H_ + j;
          float cv   = c1[ci];
          float cnew = sigm(gf) * cv + sigm(gi) * tanhf(gg);
          float hnew = sigm(go) * tanhf(cnew);
          c1[ci] = cnew;
          h1[(size_t)(par ^ 1) * BH + ci] = (bf16)hnew;
        }
        __builtin_amdgcn_wave_barrier();
      }
    }
    gridsync(bar_cnt, bar_gen, lgen);

    // ---------------- logits + log-softmax + NLL (16 waves) ----------------
    if (wid < 16) {
      const bf16* A = h1 + (size_t)(par ^ 1) * BH + (size_t)wid * 16 * H_;
      f32x8 acc[8];
#pragma unroll
      for (int nt = 0; nt < 8; ++nt) acc[nt] = fzero;
      for (int k = 0; k < H_; k += 32) {
        bf16x16 af = frag_a(A + k, H_, lane);
#pragma unroll
        for (int nt = 0; nt < 8; ++nt)
          acc[nt] = wmma_bf16(af, frag_b(Wlin + (size_t)(nt * 16) * H_ + k, H_, lane),
                              acc[nt]);
      }
      float bl[8];
#pragma unroll
      for (int nt = 0; nt < 8; ++nt) bl[nt] = blin[nt * 16 + cn];
#pragma unroll
      for (int r = 0; r < 8; ++r) {
        const int brow = wid * 16 + r + 8 * hi;   // batch row for this half-wave
        float vals[8];
        float mx = -3.0e38f;
#pragma unroll
        for (int nt = 0; nt < 8; ++nt) { vals[nt] = acc[nt][r] + bl[nt]; mx = fmaxf(mx, vals[nt]); }
#pragma unroll
        for (int m2 = 1; m2 < 16; m2 <<= 1) mx = fmaxf(mx, __shfl_xor(mx, m2, 16));
        float s = 0.f;
#pragma unroll
        for (int nt = 0; nt < 8; ++nt) s += __expf(vals[nt] - mx);
#pragma unroll
        for (int m2 = 1; m2 < 16; m2 <<= 1) s += __shfl_xor(s, m2, 16);
        const int tgt = z[brow * T_ + t];
        const int tt = tgt >> 4, tc = tgt & 15;
        float tv = vals[0];
#pragma unroll
        for (int nt = 1; nt < 8; ++nt) tv = (tt == nt) ? vals[nt] : tv;
        float tvb = __shfl(tv, (lane & 16) + tc, 32);   // biased target logit
        if (cn == 0) {
          float nllv = mx + __logf(s) - tvb;
          if (t < numf[brow]) nllacc[brow] += nllv;     // row owned by this lane
        }
      }
    }
    gridsync(bar_cnt, bar_gen, lgen);
  }

  if (blockIdx.x == 0) out[tid] = nllacc[tid] * (1.0f / (float)T_);
}

// ---------------------------------------------------------------------------
extern "C" void kernel_launch(void* const* d_in, const int* in_sizes, int n_in,
                              void* d_out, int out_size, void* d_ws, size_t ws_size,
                              hipStream_t stream) {
  (void)in_sizes; (void)n_in; (void)out_size; (void)ws_size;
  const float* W_ih0 = (const float*)d_in[0];
  const float* W_hh0 = (const float*)d_in[1];
  const float* b_ih0 = (const float*)d_in[2];
  const float* b_hh0 = (const float*)d_in[3];
  const float* W_ih1 = (const float*)d_in[4];
  const float* W_hh1 = (const float*)d_in[5];
  const float* b_ih1 = (const float*)d_in[6];
  const float* b_hh1 = (const float*)d_in[7];
  const float* W_lin = (const float*)d_in[8];
  const float* b_lin = (const float*)d_in[9];
  const int*   z     = (const int*)d_in[10];
  const int*   numf  = (const int*)d_in[11];
  float* out = (float*)d_out;

  char* base = (char*)d_ws;
  size_t o = 0;
  auto up = [](size_t x) { return (x + 255) & ~(size_t)255; };

  bf16*  Wr0   = (bf16*) (base + o); o = up(o + (size_t)G4H * H_ * 2);
  bf16*  Wr1   = (bf16*) (base + o); o = up(o + (size_t)G4H * 1024 * 2);
  bf16*  Wlin  = (bf16*) (base + o); o = up(o + (size_t)V_ * H_ * 2);
  float* emb0  = (float*)(base + o); o = up(o + (size_t)(V_ + 1) * G4H * 4);
  float* bias1 = (float*)(base + o); o = up(o + (size_t)G4H * 4);
  size_t stateOff = o;
  bf16*  h0    = (bf16*) (base + o); o = up(o + (size_t)2 * B_ * H_ * 2);
  bf16*  h1    = (bf16*) (base + o); o = up(o + (size_t)2 * B_ * H_ * 2);
  float* c0    = (float*)(base + o); o = up(o + (size_t)B_ * H_ * 4);
  float* c1    = (float*)(base + o); o = up(o + (size_t)B_ * H_ * 4);
  float* nll   = (float*)(base + o); o = up(o + (size_t)B_ * 4);
  int*   bar   = (int*)  (base + o); o = up(o + 2 * 4);
  const int stateInts = (int)((o - stateOff) / 4);

  // prepass: zero state + reorder/convert weights
  k_zero32  <<<(stateInts + 255) / 256, 256, 0, stream>>>((int*)(base + stateOff), stateInts);
  k_prep_wr0<<<(G4H * H_)   / 256, 256, 0, stream>>>(W_hh0, Wr0);
  k_prep_wr1<<<(G4H * 1024) / 256, 256, 0, stream>>>(W_ih1, W_hh1, Wr1);
  k_prep_emb<<<((V_ + 1) * G4H + 255) / 256, 256, 0, stream>>>(W_ih0, b_ih0, b_hh0, emb0);
  k_prep_b1 <<<G4H / 256, 256, 0, stream>>>(b_ih1, b_hh1, bias1);
  k_prep_wlin<<<(V_ * H_) / 256, 256, 0, stream>>>(W_lin, Wlin);

  lstm_persistent<<<NBLK, NTHR, 0, stream>>>(
      Wr0, Wr1, Wlin, emb0, bias1, b_lin, z, numf,
      h0, h1, c0, c1, nll, bar, bar + 1, out);
}
